// SlotAttentionAutoEncoder_34479997452850
// MI455X (gfx1250) — matrix operand order repaired and checked by reading, not compiled, not run on hardware
//
#include <hip/hip_runtime.h>
#include <hip/hip_bf16.h>
#include <math.h>

typedef __attribute__((ext_vector_type(16))) _Float16 v16h;
typedef __attribute__((ext_vector_type(8)))  _Float16 v8h;
typedef __attribute__((ext_vector_type(8)))  float    v8f;
typedef __attribute__((ext_vector_type(8)))  unsigned v8u;

#define HID    64
#define NSLOT  7
#define BATCH  32
#define HH     64
#define WWW    64
#define HWPIX  4096
#define ENC_M  (BATCH * HWPIX)          // 131072
#define DEC_M  (BATCH * NSLOT * HWPIX)  // 917504
#define NROWS  (BATCH * NSLOT)          // 224
#define AS_BYTES 24576                  // 2 x (128 rows x 48 f16) double-buffered A tiles

// =====================================================================
// Implicit-GEMM convolution / GEMM via V_WMMA_F32_16X16X32_F16.
//   Y[m][n] = act( sum_k A[m][k] * W[k][n] + bias[n] )
// A gathered on the fly from NHWC (k = (ky*KW+kx)*Cin + c).
// KH=KW=1, H=W=1, Nimg=M, Cin=K  ==> plain row-major GEMM.
// Whole weight matrix ([Kround][64] f16, pair-interleaved, zero-padded)
// is DMA'd into LDS once via GLOBAL_LOAD_ASYNC_TO_LDS_B128 and stays
// resident.  A tiles are double-buffered and software-pipelined:
// next-chunk global loads are issued before the current chunk's WMMAs,
// one barrier per chunk.  Tile: 128 rows x 64 cols per 8-wave block.
// =====================================================================
template<int KH, int KW>
__global__ __launch_bounds__(256)
void conv_wmma(const _Float16* __restrict__ X,
               const _Float16* __restrict__ Wk,   // pair-interleaved [Kround/2][64][2]
               const float*    __restrict__ bias, // [Cout] or nullptr
               _Float16*       __restrict__ Yh,   // f16 out (stride Cout) or nullptr
               float*          __restrict__ Yf,   // f32 out (stride Cout) or nullptr
               int Nimg, int H, int W, int Cin, int Cout, int pad, int relu,
               int Kround)
{
  extern __shared__ char smem[];                       // AS_BYTES + Kround*128
  _Float16 (*As)[128][48] = (_Float16(*)[128][48])smem;
  const unsigned* Wl32 = (const unsigned*)(smem + AS_BYTES);

  const int tid  = threadIdx.x;
  const int wave = tid >> 5;
  const int lane = tid & 31;
  const int half = lane >> 4;
  const int lm   = lane & 15;

  const long M      = (long)Nimg * H * W;
  const long m_base = (long)blockIdx.x * 128;
  const int  Ktot   = KH * KW * Cin;
  const int  nCh    = Kround >> 5;

  // ---- async DMA: whole weight matrix -> LDS (ASYNCcnt tracked) ----
  {
    const unsigned Welems = (unsigned)Kround << 6;     // f16 elements
    unsigned i    = (unsigned)tid * 8u;
    unsigned lds0 = (unsigned)(unsigned long long)(smem + AS_BYTES) + i * 2u;
    const _Float16* g = Wk + i;
    for (; i < Welems; i += 2048u, lds0 += 4096u, g += 2048u) {
      asm volatile("global_load_async_to_lds_b128 %0, %1, off"
                   :: "v"(lds0), "v"(g) : "memory");
    }
  }

  // staging-row pixel coords for this thread (16 A elems per thread)
  const int arow = tid >> 1;
  const int kk0  = (tid & 1) * 16;
  const long mrow = m_base + arow;
  const bool mok  = (mrow < M);
  int img = 0, py = 0, px = 0;
  if (mok) {
    long hw = (long)H * W;
    img = (int)(mrow / hw);
    int r = (int)(mrow % hw);
    py = r / W; px = r % W;
  }

  v8f acc[4];
  const v8f vz = {0.f,0.f,0.f,0.f,0.f,0.f,0.f,0.f};
  acc[0] = vz; acc[1] = vz; acc[2] = vz; acc[3] = vz;

  // wait for weight DMA before first use, and before first A barrier
  asm volatile("s_wait_asynccnt 0x0" ::: "memory");

  if (Cin == 64) {
    // =========== fast path: Cin==64, vector gathers, pipelined ===========
    // 16 consecutive k = 16 consecutive channels at one (ky,kx)
    auto gather = [&](int kc, v8h& a0, v8h& a1) {
      const int k0 = kc + kk0;
      const int c0 = k0 & 63;
      const int rr = k0 >> 6;
      const int kx = rr % KW;
      const int ky = rr / KW;
      const int iy = py + ky - pad;
      const int ix = px + kx - pad;
      a0 = v8h{}; a1 = v8h{};
      if (mok && k0 < Ktot && iy >= 0 && iy < H && ix >= 0 && ix < W) {
        const v8h* gp = (const v8h*)(X + (((long)img * H + iy) * W + ix) * 64 + c0);
        a0 = gp[0]; a1 = gp[1];
      }
    };
    // prologue: stage chunk 0
    {
      v8h a0, a1;
      gather(0, a0, a1);
      *(v8h*)&As[0][arow][kk0]     = a0;
      *(v8h*)&As[0][arow][kk0 + 8] = a1;
    }
    __syncthreads();
    for (int ch = 0; ch < nCh; ++ch) {
      const int p = ch & 1;
      const bool more = (ch + 1 < nCh);
      // issue next chunk's global loads early (overlap with WMMA below)
      v8h a0, a1;
      if (more) gather((ch + 1) << 5, a0, a1);

      // ---- A fragment from As[p], contiguous k = 16*half + s layout ----
      const unsigned* ar = (const unsigned*)&As[p][(wave << 4) + lm][0];
      v8u au;
      #pragma unroll
      for (int vv = 0; vv < 8; ++vv) {
        const int pi = 8*half + vv;
        au[vv] = ar[pi];
      }
      const v16h a = __builtin_bit_cast(v16h, au);
      // ---- 4 WMMA tiles along N; B pairs from resident LDS ----
      const int kcp = ch << 4;
      #pragma unroll
      for (int nt = 0; nt < 4; ++nt) {
        const int n = nt*16 + lm;
        v8u bu;
        #pragma unroll
        for (int vv = 0; vv < 8; ++vv) {
          bu[vv] = Wl32[(unsigned)(kcp + 8*half + vv) * 64u + n];
        }
        const v16h b = __builtin_bit_cast(v16h, bu);
        acc[nt] = __builtin_amdgcn_wmma_f32_16x16x32_f16(
            false, a, false, b, (short)0, acc[nt], false, false);
      }
      // store next tile into the alternate buffer, single barrier
      if (more) {
        *(v8h*)&As[1 - p][arow][kk0]     = a0;
        *(v8h*)&As[1 - p][arow][kk0 + 8] = a1;
      }
      __syncthreads();
    }
  } else {
    // =========== generic path (first conv layer, Cin=4) ===========
    __syncthreads();
    for (int ch = 0; ch < nCh; ++ch) {
      const int kc = ch << 5;
      #pragma unroll
      for (int e = 0; e < 16; ++e) {
        const int kk = kk0 + e;
        const int k  = kc + kk;
        _Float16 v = (_Float16)0.f;
        if (mok && k < Ktot) {
          const int c  = k % Cin;
          const int rr = k / Cin;
          const int kx = rr % KW;
          const int ky = rr / KW;
          const int iy = py + ky - pad;
          const int ix = px + kx - pad;
          if (iy >= 0 && iy < H && ix >= 0 && ix < W)
            v = X[(((long)img * H + iy) * W + ix) * Cin + c];
        }
        As[0][arow][kk] = v;
      }
      __syncthreads();
      const unsigned* ar = (const unsigned*)&As[0][(wave << 4) + lm][0];
      v8u au;
      #pragma unroll
      for (int vv = 0; vv < 8; ++vv) {
        const int pi = 8*half + vv;
        au[vv] = ar[pi];
      }
      const v16h a = __builtin_bit_cast(v16h, au);
      const int kcp = kc >> 1;
      #pragma unroll
      for (int nt = 0; nt < 4; ++nt) {
        const int n = nt*16 + lm;
        v8u bu;
        #pragma unroll
        for (int vv = 0; vv < 8; ++vv) {
          bu[vv] = Wl32[(unsigned)(kcp + 8*half + vv) * 64u + n];
        }
        const v16h b = __builtin_bit_cast(v16h, bu);
        acc[nt] = __builtin_amdgcn_wmma_f32_16x16x32_f16(
            false, a, false, b, (short)0, acc[nt], false, false);
      }
      __syncthreads();
    }
  }

  // ---- epilogue: D layout -> (m = r + 8*half, n = lane%16) ----
  #pragma unroll
  for (int nt = 0; nt < 4; ++nt) {
    const int n = nt*16 + lm;
    #pragma unroll
    for (int r = 0; r < 8; ++r) {
      const long m = m_base + wave*16 + r + 8*half;
      if (m < M && n < Cout) {
        float v = acc[nt][r];
        if (bias) v += bias[n];
        if (relu) v = v > 0.f ? v : 0.f;
        if (Yh) Yh[m * Cout + n] = (_Float16)v;
        if (Yf) Yf[m * Cout + n] = v;
      }
    }
  }
}

// ============================ helpers ================================

// OIHW f32 -> pair-interleaved [Kround/2][64][2] f16, zero-padded.
// flipT: convT repack (orig (I,O,KH,KW)): w2[o][c][ky][kx] = orig[c][o][KH-1-ky][KW-1-kx]
__global__ void k_repack_conv(const float* __restrict__ w, _Float16* __restrict__ out,
                              int O, int I, int KHp, int KWp, int Cpad, int flipT,
                              int Kround)
{
  const int Ktot = KHp * KWp * Cpad;
  const int idx = blockIdx.x * blockDim.x + threadIdx.x;
  if (idx >= Kround * 64) return;
  const int n = idx & 63;
  const int k = idx >> 6;
  float v = 0.f;
  if (k < Ktot) {
    const int c = k % Cpad;
    const int rr = k / Cpad;
    const int kx = rr % KWp;
    const int ky = rr / KWp;
    if (n < O && c < I) {
      if (!flipT) v = w[(((long)n * I + c) * KHp + ky) * KWp + kx];
      else        v = w[(((long)c * O + n) * KHp + (KHp-1-ky)) * KWp + (KWp-1-kx)];
    }
  }
  out[((k >> 1) * 64 + n) * 2 + (k & 1)] = (_Float16)v;
}

// linear weight (N,64) f32 -> pair-interleaved [32][64][2] f16
__global__ void k_repack_lin(const float* __restrict__ w, _Float16* __restrict__ out, int N)
{
  const int idx = blockIdx.x * blockDim.x + threadIdx.x;
  if (idx >= 64 * 64) return;
  const int n = idx & 63;
  const int k = idx >> 6;
  const float v = (n < N) ? w[(long)n * 64 + k] : 0.f;
  out[((k >> 1) * 64 + n) * 2 + (k & 1)] = (_Float16)v;
}

// positional embedding: pe[pix][d] = b[d] + W[d][:] . [y,x,1-y,1-x]
__global__ void k_pos_embed(const float* __restrict__ w, const float* __restrict__ b,
                            float* __restrict__ pe)
{
  const int idx = blockIdx.x * blockDim.x + threadIdx.x;
  if (idx >= HWPIX * HID) return;
  const int d = idx & 63;
  const int pix = idx >> 6;
  const float gy = (float)(pix >> 6) * (1.f / 63.f);
  const float gx = (float)(pix & 63) * (1.f / 63.f);
  pe[idx] = b[d] + w[d*4+0]*gy + w[d*4+1]*gx + w[d*4+2]*(1.f-gy) + w[d*4+3]*(1.f-gx);
}

// image NCHW f32 -> NHWC f16 with C padded 3->4
__global__ void k_prep_input(const float* __restrict__ img, _Float16* __restrict__ out)
{
  const long idx = (long)blockIdx.x * blockDim.x + threadIdx.x;
  if (idx >= (long)ENC_M * 4) return;
  const int c = (int)(idx & 3);
  const long m = idx >> 2;
  const int n = (int)(m >> 12);
  const int pix = (int)(m & 4095);
  float v = 0.f;
  if (c < 3) v = img[(((long)n * 3 + c) * 64 + (pix >> 6)) * 64 + (pix & 63)];
  out[idx] = (_Float16)v;
}

__global__ void k_add_pe(const _Float16* __restrict__ x, const float* __restrict__ pe,
                         _Float16* __restrict__ y, long total)
{
  const long idx = (long)blockIdx.x * blockDim.x + threadIdx.x;
  if (idx >= total) return;
  const int d = (int)(idx & 63);
  const int pix = (int)((idx >> 6) & 4095);
  y[idx] = (_Float16)((float)x[idx] + pe[pix * 64 + d]);
}

// per-batch mean/var over HW*HID elements
__global__ void k_batch_stats(const _Float16* __restrict__ x, float* __restrict__ stat, int perB)
{
  __shared__ float ss[256], sq[256];
  const int b = blockIdx.x;
  const _Float16* p = x + (long)b * perB;
  float s = 0.f, q = 0.f;
  for (int i = threadIdx.x; i < perB; i += 256) {
    float v = (float)p[i];
    s += v; q += v * v;
  }
  ss[threadIdx.x] = s; sq[threadIdx.x] = q;
  __syncthreads();
  for (int o = 128; o > 0; o >>= 1) {
    if (threadIdx.x < o) { ss[threadIdx.x] += ss[threadIdx.x+o]; sq[threadIdx.x] += sq[threadIdx.x+o]; }
    __syncthreads();
  }
  if (threadIdx.x == 0) {
    float m = ss[0] / perB;
    stat[b*2] = m;
    stat[b*2+1] = sq[0] / perB - m * m;
  }
}

__global__ void k_batch_norm(const _Float16* __restrict__ x, const float* __restrict__ stat,
                             _Float16* __restrict__ y, long total)
{
  const long idx = (long)blockIdx.x * blockDim.x + threadIdx.x;
  if (idx >= total) return;
  const int b = (int)(idx >> 18);            // 4096*64 elems per batch
  const float m = stat[b*2];
  const float inv = rsqrtf(stat[b*2+1] + 1e-5f);
  y[idx] = (_Float16)(((float)x[idx] - m) * inv);
}

// LayerNorm over last dim (64); one wave per row; input f16 OR f32
__global__ void k_ln(const _Float16* __restrict__ in16, const float* __restrict__ in32,
                     const float* __restrict__ g, const float* __restrict__ bb,
                     _Float16* __restrict__ out16, long M)
{
  const int wave = threadIdx.x >> 5;
  const int lane = threadIdx.x & 31;
  const long row = (long)blockIdx.x * 8 + wave;
  if (row >= M) return;
  const long base = row * 64;
  float x0 = in32 ? in32[base + lane]      : (float)in16[base + lane];
  float x1 = in32 ? in32[base + lane + 32] : (float)in16[base + lane + 32];
  float s = x0 + x1;
  #pragma unroll
  for (int o = 16; o > 0; o >>= 1) s += __shfl_xor(s, o, 32);
  const float mean = s * (1.f / 64.f);
  const float d0 = x0 - mean, d1 = x1 - mean;
  float vs = d0*d0 + d1*d1;
  #pragma unroll
  for (int o = 16; o > 0; o >>= 1) vs += __shfl_xor(vs, o, 32);
  const float inv = rsqrtf(vs * (1.f / 64.f) + 1e-5f);
  out16[base + lane]      = (_Float16)(d0 * inv * g[lane]    + bb[lane]);
  out16[base + lane + 32] = (_Float16)(d1 * inv * g[lane+32] + bb[lane+32]);
}

// deterministic hash-gaussian slot init: slots = mu + sigma * N(0,1)
__device__ __forceinline__ unsigned h32(unsigned x) {
  x = x * 747796405u + 2891336453u;
  x = ((x >> ((x >> 28) + 4u)) ^ x) * 277803737u;
  return x ^ (x >> 22);
}
__global__ void k_slot_init(const float* __restrict__ mu, const float* __restrict__ sigma,
                            float* __restrict__ slots)
{
  const int idx = blockIdx.x * blockDim.x + threadIdx.x;
  if (idx >= NROWS * HID) return;
  const int d = idx & 63;
  const float u1 = (float)((h32(idx) & 0xFFFFFFu)) * (1.f/16777216.f) + 1e-7f;
  const float u2 = (float)((h32(idx ^ 0x9e3779b9u) & 0xFFFFFFu)) * (1.f/16777216.f);
  const float n = sqrtf(-2.f * logf(u1)) * cosf(6.28318530718f * u2);
  slots[idx] = mu[d] + sigma[d] * n;
}

// dots = scale * q.k ; softmax over slot axis (7) ; +EPS ; store unnormalized
__global__ void k_attn_dots(const _Float16* __restrict__ q, const _Float16* __restrict__ kk,
                            float* __restrict__ attn_un)
{
  __shared__ float qs[NSLOT * HID];
  const int b = blockIdx.x >> 4;
  const int j = (blockIdx.x & 15) * 256 + threadIdx.x;
  for (int i = threadIdx.x; i < NSLOT * HID; i += 256)
    qs[i] = (float)q[(long)b * NSLOT * HID + i];
  __syncthreads();
  float acc[NSLOT];
  #pragma unroll
  for (int i = 0; i < NSLOT; ++i) acc[i] = 0.f;
  const long kb = ((long)b * HWPIX + j) * HID;
  for (int d = 0; d < HID; ++d) {
    const float kd = (float)kk[kb + d];
    #pragma unroll
    for (int i = 0; i < NSLOT; ++i) acc[i] += qs[i * HID + d] * kd;
  }
  float mx = -1e30f;
  #pragma unroll
  for (int i = 0; i < NSLOT; ++i) { acc[i] *= 0.125f; mx = fmaxf(mx, acc[i]); }
  float se = 0.f;
  #pragma unroll
  for (int i = 0; i < NSLOT; ++i) { acc[i] = expf(acc[i] - mx); se += acc[i]; }
  const float rinv = 1.f / se;
  #pragma unroll
  for (int i = 0; i < NSLOT; ++i)
    attn_un[((long)(b * NSLOT + i)) * HWPIX + j] = acc[i] * rinv + 1e-8f;
}

__global__ void k_attn_colsum(const float* __restrict__ attn_un, float* __restrict__ sums)
{
  __shared__ float red[256];
  const int bi = blockIdx.x;
  float s = 0.f;
  for (int j = threadIdx.x; j < HWPIX; j += 256) s += attn_un[(long)bi * HWPIX + j];
  red[threadIdx.x] = s;
  __syncthreads();
  for (int o = 128; o > 0; o >>= 1) {
    if (threadIdx.x < o) red[threadIdx.x] += red[threadIdx.x + o];
    __syncthreads();
  }
  if (threadIdx.x == 0) sums[bi] = red[0];
}

__global__ void k_attn_updates(const _Float16* __restrict__ v, const float* __restrict__ attn_un,
                               const float* __restrict__ sums, float* __restrict__ upd)
{
  const int bi = blockIdx.x;
  const int b = bi / NSLOT;
  const int d = threadIdx.x;
  float acc = 0.f;
  for (int j = 0; j < HWPIX; ++j)
    acc += (float)v[((long)b * HWPIX + j) * HID + d] * attn_un[(long)bi * HWPIX + j];
  upd[bi * HID + d] = acc / sums[bi];
}

__global__ void k_gru(const float* __restrict__ upd, float* __restrict__ slots,
                      const float* __restrict__ wih, const float* __restrict__ bih,
                      const float* __restrict__ whh, const float* __restrict__ bhh)
{
  __shared__ float xs[HID], hs[HID];
  const int r = blockIdx.x;
  const int d = threadIdx.x;
  xs[d] = upd[r * HID + d];
  hs[d] = slots[r * HID + d];
  __syncthreads();
  float gi[3], gh[3];
  #pragma unroll
  for (int g = 0; g < 3; ++g) {
    const int o = g * HID + d;
    float si = bih[o], sh = bhh[o];
    for (int k = 0; k < HID; ++k) {
      si += xs[k] * wih[(long)o * HID + k];
      sh += hs[k] * whh[(long)o * HID + k];
    }
    gi[g] = si; gh[g] = sh;
  }
  const float rg = 1.f / (1.f + expf(-(gi[0] + gh[0])));
  const float zg = 1.f / (1.f + expf(-(gi[1] + gh[1])));
  const float ng = tanhf(gi[2] + rg * gh[2]);
  const float hn = (1.f - zg) * ng + zg * hs[d];
  __syncthreads();
  slots[r * HID + d] = hn;
}

__global__ void k_add_f32(float* __restrict__ a, const float* __restrict__ b, int n)
{
  const int i = blockIdx.x * blockDim.x + threadIdx.x;
  if (i < n) a[i] += b[i];
}

__global__ void k_write_attn(const float* __restrict__ attn_un, const float* __restrict__ sums,
                             float* __restrict__ out)
{
  const long idx = (long)blockIdx.x * blockDim.x + threadIdx.x;
  if (idx >= (long)NROWS * HWPIX) return;
  const int bi = (int)(idx >> 12);
  out[idx] = attn_un[idx] / sums[bi];
}

__global__ void k_copy_f32(const float* __restrict__ a, float* __restrict__ b, int n)
{
  const int i = blockIdx.x * blockDim.x + threadIdx.x;
  if (i < n) b[i] = a[i];
}

__global__ void k_build_dec(const float* __restrict__ slots, const float* __restrict__ pe,
                            _Float16* __restrict__ z)
{
  const long idx = (long)blockIdx.x * blockDim.x + threadIdx.x;
  if (idx >= (long)DEC_M * HID) return;
  const int d = (int)(idx & 63);
  const long m = idx >> 6;
  const int s = (int)(m >> 12);
  const int pix = (int)(m & 4095);
  z[idx] = (_Float16)(slots[s * HID + d] + pe[pix * HID + d]);
}

// per-pixel softmax over slots, combine, write 3 outputs
__global__ void k_combine(const float* __restrict__ z2, float* __restrict__ out,
                          long O_RECONS, long O_MASKS)
{
  const int idx = blockIdx.x * blockDim.x + threadIdx.x;
  if (idx >= ENC_M) return;
  const int b = idx >> 12;
  const int pix = idx & 4095;
  float rec[NSLOT], ml[NSLOT];
  #pragma unroll
  for (int s = 0; s < NSLOT; ++s) {
    const long zi = (((long)(b * NSLOT + s)) * HWPIX + pix) * 2;
    rec[s] = z2[zi];
    ml[s]  = z2[zi + 1];
  }
  float mx = -1e30f;
  #pragma unroll
  for (int s = 0; s < NSLOT; ++s) mx = fmaxf(mx, ml[s]);
  float se = 0.f;
  #pragma unroll
  for (int s = 0; s < NSLOT; ++s) { ml[s] = expf(ml[s] - mx); se += ml[s]; }
  const float rinv = 1.f / se;
  float comb = 0.f;
  #pragma unroll
  for (int s = 0; s < NSLOT; ++s) {
    const float msk = ml[s] * rinv;
    const long oi = (long)(b * NSLOT + s) * HWPIX + pix;
    out[O_RECONS + oi] = rec[s];
    out[O_MASKS + oi]  = msk;
    comb += rec[s] * msk;
  }
  out[idx] = comb;  // recon_combined at offset 0
}

// ============================== host =================================
static inline unsigned nblk(long n, int t) { return (unsigned)((n + t - 1) / t); }
static inline int kround32(int k) { return (k + 31) & ~31; }

extern "C" void kernel_launch(void* const* d_in, const int* in_sizes, int n_in,
                              void* d_out, int out_size, void* d_ws, size_t ws_size,
                              hipStream_t stream)
{
  (void)in_sizes; (void)n_in; (void)out_size; (void)ws_size;
  const float* image    = (const float*)d_in[0];
  const float* enc_w1   = (const float*)d_in[1];
  const float* enc_b1   = (const float*)d_in[2];
  const float* enc_w2   = (const float*)d_in[3];
  const float* enc_b2   = (const float*)d_in[4];
  const float* enc_w3   = (const float*)d_in[5];
  const float* enc_b3   = (const float*)d_in[6];
  const float* enc_w4   = (const float*)d_in[7];
  const float* enc_b4   = (const float*)d_in[8];
  const float* enc_posw = (const float*)d_in[9];
  const float* enc_posb = (const float*)d_in[10];
  const float* dec_posw = (const float*)d_in[11];
  const float* dec_posb = (const float*)d_in[12];
  const float* dec_w1   = (const float*)d_in[13];
  const float* dec_b1   = (const float*)d_in[14];
  const float* dec_w2   = (const float*)d_in[15];
  const float* dec_b2   = (const float*)d_in[16];
  const float* dec_w3   = (const float*)d_in[17];
  const float* dec_b3   = (const float*)d_in[18];
  const float* dec_w4   = (const float*)d_in[19];
  const float* dec_b4   = (const float*)d_in[20];
  const float* fc1_w    = (const float*)d_in[21];
  const float* fc1_b    = (const float*)d_in[22];
  const float* fc2_w    = (const float*)d_in[23];
  const float* fc2_b    = (const float*)d_in[24];
  const float* mu       = (const float*)d_in[25];
  const float* sigma    = (const float*)d_in[26];
  const float* q_w      = (const float*)d_in[27];
  const float* q_b      = (const float*)d_in[28];
  const float* k_w      = (const float*)d_in[29];
  const float* k_b      = (const float*)d_in[30];
  const float* v_w      = (const float*)d_in[31];
  const float* v_b      = (const float*)d_in[32];
  const float* gru_wih  = (const float*)d_in[33];
  const float* gru_bih  = (const float*)d_in[34];
  const float* gru_whh  = (const float*)d_in[35];
  const float* gru_bhh  = (const float*)d_in[36];
  const float* sfc1_w   = (const float*)d_in[37];
  const float* sfc1_b   = (const float*)d_in[38];
  const float* sfc2_w   = (const float*)d_in[39];
  const float* sfc2_b   = (const float*)d_in[40];
  const float* ln_in_g  = (const float*)d_in[41];
  const float* ln_in_b  = (const float*)d_in[42];
  const float* ln_sl_g  = (const float*)d_in[43];
  const float* ln_sl_b  = (const float*)d_in[44];
  const float* ln_ff_g  = (const float*)d_in[45];
  const float* ln_ff_b  = (const float*)d_in[46];

  float* out = (float*)d_out;
  const long O_RECONS = 131072;
  const long O_MASKS  = O_RECONS + 917504;
  const long O_SLOTS  = O_MASKS + 917504;
  const long O_ATTN   = O_SLOTS + 14336;

  const int KR1  = kround32(25 * 4);    // 128  (enc conv1, Cin padded to 4)
  const int KR25 = kround32(25 * 64);   // 1600
  const int KR9  = kround32(9 * 64);    // 576
  const int KR64 = 64;                  // 1x1 GEMMs

  // ---- carve workspace ----
  char* wsb = (char*)d_ws;
  size_t off = 0;
  auto carve = [&](size_t bytes) -> void* {
    void* p = wsb + off;
    off = (off + bytes + 255) & ~(size_t)255;
    return p;
  };
  _Float16* bufA   = (_Float16*)carve((size_t)DEC_M * HID * 2);
  _Float16* bufB   = (_Float16*)carve((size_t)DEC_M * HID * 2);
  _Float16* kbuf   = (_Float16*)carve((size_t)ENC_M * HID * 2);
  _Float16* vbuf   = (_Float16*)carve((size_t)ENC_M * HID * 2);
  _Float16* qbuf   = (_Float16*)carve((size_t)NROWS * HID * 2);
  _Float16* ln16   = (_Float16*)carve((size_t)NROWS * HID * 2);
  _Float16* g1f16  = (_Float16*)carve((size_t)NROWS * HID * 2);
  float*    attnun = (float*)carve((size_t)NROWS * HWPIX * 4);
  float*    attnsm = (float*)carve((size_t)NROWS * 4);
  float*    slots  = (float*)carve((size_t)NROWS * HID * 4);
  float*    upd    = (float*)carve((size_t)NROWS * HID * 4);
  float*    fff32  = (float*)carve((size_t)NROWS * HID * 4);
  float*    peE    = (float*)carve((size_t)HWPIX * HID * 4);
  float*    peD    = (float*)carve((size_t)HWPIX * HID * 4);
  float*    bstat  = (float*)carve((size_t)BATCH * 2 * 4);
  _Float16* xpad   = (_Float16*)carve((size_t)ENC_M * 4 * 2);
  float*    zout   = (float*)carve((size_t)DEC_M * 2 * 4);
  _Float16* wpE1   = (_Float16*)carve((size_t)KR1  * 64 * 2);
  _Float16* wpE2   = (_Float16*)carve((size_t)KR25 * 64 * 2);
  _Float16* wpE3   = (_Float16*)carve((size_t)KR25 * 64 * 2);
  _Float16* wpE4   = (_Float16*)carve((size_t)KR25 * 64 * 2);
  _Float16* wpD1   = (_Float16*)carve((size_t)KR25 * 64 * 2);
  _Float16* wpD2   = (_Float16*)carve((size_t)KR25 * 64 * 2);
  _Float16* wpD3   = (_Float16*)carve((size_t)KR25 * 64 * 2);
  _Float16* wpD4   = (_Float16*)carve((size_t)KR9  * 64 * 2);
  _Float16* wpFC1  = (_Float16*)carve((size_t)64 * 64 * 2);
  _Float16* wpFC2  = (_Float16*)carve((size_t)64 * 64 * 2);
  _Float16* wpQ    = (_Float16*)carve((size_t)64 * 64 * 2);
  _Float16* wpK    = (_Float16*)carve((size_t)64 * 64 * 2);
  _Float16* wpV    = (_Float16*)carve((size_t)64 * 64 * 2);
  _Float16* wpS1   = (_Float16*)carve((size_t)64 * 64 * 2);
  _Float16* wpS2   = (_Float16*)carve((size_t)64 * 64 * 2);

  // dynamic LDS sizes: double-buffered A tiles + resident weights
  const size_t LDS1  = AS_BYTES + (size_t)KR1  * 128;
  const size_t LDS25 = AS_BYTES + (size_t)KR25 * 128;
  const size_t LDS9  = AS_BYTES + (size_t)KR9  * 128;
  const size_t LDS64 = AS_BYTES + (size_t)KR64 * 128;

  // ---- weight repacks ----
  k_repack_conv<<<nblk(KR1*64,256),256,0,stream>>>(enc_w1, wpE1, 64, 3, 5, 5, 4, 0, KR1);
  k_repack_conv<<<nblk(KR25*64,256),256,0,stream>>>(enc_w2, wpE2, 64, 64, 5, 5, 64, 0, KR25);
  k_repack_conv<<<nblk(KR25*64,256),256,0,stream>>>(enc_w3, wpE3, 64, 64, 5, 5, 64, 0, KR25);
  k_repack_conv<<<nblk(KR25*64,256),256,0,stream>>>(enc_w4, wpE4, 64, 64, 5, 5, 64, 0, KR25);
  k_repack_conv<<<nblk(KR25*64,256),256,0,stream>>>(dec_w1, wpD1, 64, 64, 5, 5, 64, 1, KR25);
  k_repack_conv<<<nblk(KR25*64,256),256,0,stream>>>(dec_w2, wpD2, 64, 64, 5, 5, 64, 1, KR25);
  k_repack_conv<<<nblk(KR25*64,256),256,0,stream>>>(dec_w3, wpD3, 64, 64, 5, 5, 64, 1, KR25);
  k_repack_conv<<<nblk(KR9*64,256),256,0,stream>>>(dec_w4, wpD4, 2, 64, 3, 3, 64, 1, KR9);
  k_repack_lin<<<16,256,0,stream>>>(fc1_w,  wpFC1, 64);
  k_repack_lin<<<16,256,0,stream>>>(fc2_w,  wpFC2, 64);
  k_repack_lin<<<16,256,0,stream>>>(q_w,    wpQ,   64);
  k_repack_lin<<<16,256,0,stream>>>(k_w,    wpK,   64);
  k_repack_lin<<<16,256,0,stream>>>(v_w,    wpV,   64);
  k_repack_lin<<<16,256,0,stream>>>(sfc1_w, wpS1,  64);
  k_repack_lin<<<16,256,0,stream>>>(sfc2_w, wpS2,  64);
  k_pos_embed<<<nblk(HWPIX*HID,256),256,0,stream>>>(enc_posw, enc_posb, peE);
  k_pos_embed<<<nblk(HWPIX*HID,256),256,0,stream>>>(dec_posw, dec_posb, peD);

  // ---- encoder ----
  k_prep_input<<<nblk((long)ENC_M*4,256),256,0,stream>>>(image, xpad);
  const unsigned gEnc = nblk(ENC_M, 128);
  conv_wmma<5,5><<<gEnc,256,LDS1,stream>>>(xpad, wpE1, enc_b1, bufA, nullptr, BATCH, HH, WWW, 4, 64, 2, 1, KR1);
  conv_wmma<5,5><<<gEnc,256,LDS25,stream>>>(bufA, wpE2, enc_b2, bufB, nullptr, BATCH, HH, WWW, 64, 64, 2, 1, KR25);
  conv_wmma<5,5><<<gEnc,256,LDS25,stream>>>(bufB, wpE3, enc_b3, bufA, nullptr, BATCH, HH, WWW, 64, 64, 2, 1, KR25);
  conv_wmma<5,5><<<gEnc,256,LDS25,stream>>>(bufA, wpE4, enc_b4, bufB, nullptr, BATCH, HH, WWW, 64, 64, 2, 1, KR25);
  k_add_pe<<<nblk((long)ENC_M*HID,256),256,0,stream>>>(bufB, peE, bufA, (long)ENC_M*HID);
  k_batch_stats<<<BATCH,256,0,stream>>>(bufA, bstat, HWPIX*HID);
  k_batch_norm<<<nblk((long)ENC_M*HID,256),256,0,stream>>>(bufA, bstat, bufB, (long)ENC_M*HID);
  conv_wmma<1,1><<<gEnc,256,LDS64,stream>>>(bufB, wpFC1, fc1_b, bufA, nullptr, ENC_M, 1, 1, 64, 64, 0, 1, KR64);
  conv_wmma<1,1><<<gEnc,256,LDS64,stream>>>(bufA, wpFC2, fc2_b, bufB, nullptr, ENC_M, 1, 1, 64, 64, 0, 0, KR64);
  k_ln<<<nblk(ENC_M,8),256,0,stream>>>(bufB, nullptr, ln_in_g, ln_in_b, bufA, ENC_M);
  conv_wmma<1,1><<<gEnc,256,LDS64,stream>>>(bufA, wpK, k_b, kbuf, nullptr, ENC_M, 1, 1, 64, 64, 0, 0, KR64);
  conv_wmma<1,1><<<gEnc,256,LDS64,stream>>>(bufA, wpV, v_b, vbuf, nullptr, ENC_M, 1, 1, 64, 64, 0, 0, KR64);

  // ---- slot attention (3 iterations) ----
  k_slot_init<<<nblk(NROWS*HID,256),256,0,stream>>>(mu, sigma, slots);
  const unsigned gRow = nblk(NROWS, 128);
  for (int it = 0; it < 3; ++it) {
    k_ln<<<nblk(NROWS,8),256,0,stream>>>(nullptr, slots, ln_sl_g, ln_sl_b, ln16, NROWS);
    conv_wmma<1,1><<<gRow,256,LDS64,stream>>>(ln16, wpQ, q_b, qbuf, nullptr, NROWS, 1, 1, 64, 64, 0, 0, KR64);
    k_attn_dots<<<BATCH*16,256,0,stream>>>(qbuf, kbuf, attnun);
    k_attn_colsum<<<NROWS,256,0,stream>>>(attnun, attnsm);
    k_attn_updates<<<NROWS,64,0,stream>>>(vbuf, attnun, attnsm, upd);
    k_gru<<<NROWS,64,0,stream>>>(upd, slots, gru_wih, gru_bih, gru_whh, gru_bhh);
    k_ln<<<nblk(NROWS,8),256,0,stream>>>(nullptr, slots, ln_ff_g, ln_ff_b, ln16, NROWS);
    conv_wmma<1,1><<<gRow,256,LDS64,stream>>>(ln16, wpS1, sfc1_b, g1f16, nullptr, NROWS, 1, 1, 64, 64, 0, 1, KR64);
    conv_wmma<1,1><<<gRow,256,LDS64,stream>>>(g1f16, wpS2, sfc2_b, nullptr, fff32, NROWS, 1, 1, 64, 64, 0, 0, KR64);
    k_add_f32<<<nblk(NROWS*HID,256),256,0,stream>>>(slots, fff32, NROWS*HID);
  }
  k_write_attn<<<nblk((long)NROWS*HWPIX,256),256,0,stream>>>(attnun, attnsm, out + O_ATTN);
  k_copy_f32<<<nblk(NROWS*HID,256),256,0,stream>>>(slots, out + O_SLOTS, NROWS*HID);

  // ---- decoder ----
  k_build_dec<<<nblk((long)DEC_M*HID,256),256,0,stream>>>(slots, peD, bufA);
  const unsigned gDec = nblk(DEC_M, 128);
  conv_wmma<5,5><<<gDec,256,LDS25,stream>>>(bufA, wpD1, dec_b1, bufB, nullptr, NROWS, HH, WWW, 64, 64, 2, 1, KR25);
  conv_wmma<5,5><<<gDec,256,LDS25,stream>>>(bufB, wpD2, dec_b2, bufA, nullptr, NROWS, HH, WWW, 64, 64, 2, 1, KR25);
  conv_wmma<5,5><<<gDec,256,LDS25,stream>>>(bufA, wpD3, dec_b3, bufB, nullptr, NROWS, HH, WWW, 64, 64, 2, 1, KR25);
  conv_wmma<3,3><<<gDec,256,LDS9,stream>>>(bufB, wpD4, dec_b4, nullptr, zout, NROWS, HH, WWW, 64, 2, 1, 0, KR9);
  k_combine<<<nblk(ENC_M,256),256,0,stream>>>(zout, out, O_RECONS, O_MASKS);
}
